// TCNN_66743791780427
// MI455X (gfx1250) — matrix operand order repaired and checked by reading, not compiled, not run on hardware
//
#include <hip/hip_runtime.h>
#include <math.h>

typedef _Float16 h16v __attribute__((ext_vector_type(16)));
typedef _Float16 h8v  __attribute__((ext_vector_type(8)));
typedef float    f8v  __attribute__((ext_vector_type(8)));

// ---------------- workspace layout (bytes) -----------------------------
// w2h : 64*800  f16 = 102,400   (K permuted: K = (ky*5+kx)*32 + ic)
// w3h : 512*1024 f16 = 1,048,576
// sc  : 1216 f32 (sc1[32] sh1[32] sc2[64] sh2[64] sc3[512] sh3[512]) = 4,864
// act2: 8192*1024 f16 = 16,777,216
// act3: 8192*512  f16 =  8,388,608    -> total ~26.3 MB
#define WS_W2H   0
#define WS_W3H   102400
#define WS_SC    1150976
#define WS_ACT2  1155840
#define WS_ACT3  17933056

// ---------------- prep: f32 -> f16 weight conversion + K-permute -------
__global__ __launch_bounds__(256) void k_prep_w(
    const float* __restrict__ w2, const float* __restrict__ w3,
    _Float16* __restrict__ w2h, _Float16* __restrict__ w3h)
{
  int i = blockIdx.x * 256 + threadIdx.x;
  if (i < 64 * 800) {
    // dest K-order (ky,kx,ic); src w2 is [oc][ic][ky][kx]
    int oc = i / 800, rem = i - oc * 800;
    int t = rem >> 5, ic = rem & 31;
    int ky = t / 5, kx = t - ky * 5;
    w2h[i] = (_Float16)w2[oc * 800 + ic * 25 + ky * 5 + kx];
  }
  if (i < 512 * 1024) w3h[i] = (_Float16)w3[i];
}

// ---------------- prep: fold bias + BN into (scale, shift) -------------
__global__ __launch_bounds__(256) void k_prep_bn(
    const float* b1, const float* g1, const float* be1, const float* m1, const float* v1,
    const float* b2, const float* g2, const float* be2, const float* m2, const float* v2,
    const float* b3, const float* g3, const float* be3, const float* m3, const float* v3,
    float* __restrict__ sc)
{
  int i = blockIdx.x * 256 + threadIdx.x;
  if (i < 32) {
    float inv = g1[i] * rsqrtf(v1[i] + 1e-5f);
    sc[i] = inv;  sc[32 + i] = b1[i] * inv + be1[i] - m1[i] * inv;
  } else if (i < 96) {
    int c = i - 32;
    float inv = g2[c] * rsqrtf(v2[c] + 1e-5f);
    sc[64 + c] = inv;  sc[128 + c] = b2[c] * inv + be2[c] - m2[c] * inv;
  } else if (i < 608) {
    int c = i - 96;
    float inv = g3[c] * rsqrtf(v3[c] + 1e-5f);
    sc[192 + c] = inv;  sc[704 + c] = b3[c] * inv + be3[c] - m3[c] * inv;
  }
}

// ---------------- fused conv1+bn+relu+pool+conv2+bn+relu+pool ----------
// one workgroup (128 thr = 4 waves) per image; all intermediates in LDS
__global__ __launch_bounds__(128) void k_conv12(
    const float* __restrict__ x, const float* __restrict__ w1,
    const _Float16* __restrict__ w2h, const float* __restrict__ sc,
    _Float16* __restrict__ act2)
{
  __shared__ _Float16 s_xh[784];              // image as f16
  __shared__ _Float16 s_w1h[32 * 32];         // conv1 weights, K padded 25->32
  __shared__ short    s_d1[32];               // conv1 im2col offsets
  __shared__ _Float16 s_a1[144 * 32];         // pooled conv1 act, channel-last [pos][ic]
  __shared__ __align__(32) char s_big[576 * 32 * 2]; // phase1: conv1 raw / phase2: conv2 raw
  _Float16* s_c1 = (_Float16*)s_big;          // [pos=576][ch=32] f16, phase 1
  float*    s_a2 = (float*)s_big;             // [n=64][ch=64] f32, phase 2

  const int tid  = threadIdx.x;
  const int img  = blockIdx.x;
  // wave id is uniform across the wave: make it scalar so wave-level loops
  // compile to s_cbranch (EXEC provably all-1s around every WMMA)
  const int wv   = __builtin_amdgcn_readfirstlane(tid >> 5);
  const int lane = tid & 31;
  const int col  = lane & 15;
  const int hi   = lane >> 4;

  // ---- phase 0: stage image, conv1 weights, gather table ----
  for (int i = tid; i < 784; i += 128) s_xh[i] = (_Float16)x[img * 784 + i];
  for (int i = tid; i < 1024; i += 128) {
    int oc = i >> 5, k = i & 31;
    s_w1h[i] = (k < 25) ? (_Float16)w1[oc * 25 + k] : (_Float16)0.0f;
  }
  if (tid < 32) s_d1[tid] = (tid < 25) ? (short)((tid / 5) * 28 + tid % 5) : (short)0;
  __syncthreads();

  // ---- phase 1a: conv1 as WMMA implicit GEMM ----
  // M=32 (2 tiles), K=32 (25 real taps), N=576 (36 tiles across 4 waves)
  for (int nt = wv; nt < 36; nt += 4) {
    int n1 = nt * 16 + col;
    int oy = n1 / 24, ox = n1 - oy * 24;
    int base1 = oy * 28 + ox;
    union { h16v v; _Float16 h[16]; } B;
#pragma unroll
    for (int j = 0; j < 16; ++j) {
      int kk = hi * 16 + j;
      B.h[j] = (kk < 25) ? s_xh[base1 + s_d1[kk]] : (_Float16)0.0f;
    }
    f8v z = {0.f, 0.f, 0.f, 0.f, 0.f, 0.f, 0.f, 0.f};
#pragma unroll
    for (int mt = 0; mt < 2; ++mt) {
      union { h16v v; h8v p[2]; } A;
      const _Float16* wr = s_w1h + (mt * 16 + col) * 32 + hi * 16;
      A.p[0] = *(const h8v*)wr;
      A.p[1] = *(const h8v*)(wr + 8);
      union { f8v v; float f[8]; } C;
      C.v = __builtin_amdgcn_wmma_f32_16x16x32_f16(false, A.v, false, B.v,
                                                   (short)0, z, false, false);
      // bn1 + relu -> channel-last [pos][ch]: 8 contiguous halves per lane
      union { h8v v; _Float16 h[8]; } o;
#pragma unroll
      for (int r = 0; r < 8; ++r) {
        int ch = mt * 16 + hi * 8 + r;
        float v = C.f[r] * sc[ch] + sc[32 + ch];
        o.h[r] = (_Float16)(v > 0.f ? v : 0.f);
      }
      *(h8v*)(s_c1 + n1 * 32 + mt * 16 + hi * 8) = o.v;
    }
  }
  __syncthreads();

  // ---- phase 1b: 2x2 maxpool, vectorized 8 ch at a time -> s_a1 [144][32]
  for (int o = tid; o < 576; o += 128) {
    int p = o >> 2, g = o & 3;              // p: pooled pos, g: 8-channel group
    int py = p / 12, px = p - py * 12;
    int cb = g * 8;
    const _Float16* c00 = s_c1 + ((2 * py) * 24 + 2 * px) * 32 + cb;
    union { h8v v; _Float16 h[8]; } a, b, c, d, m;
    a.v = *(const h8v*)c00;
    b.v = *(const h8v*)(c00 + 32);          // x+1
    c.v = *(const h8v*)(c00 + 24 * 32);     // y+1
    d.v = *(const h8v*)(c00 + 25 * 32);     // x+1, y+1
#pragma unroll
    for (int r = 0; r < 8; ++r) {
      float mm = fmaxf(fmaxf((float)a.h[r], (float)b.h[r]),
                       fmaxf((float)c.h[r], (float)d.h[r]));
      m.h[r] = (_Float16)mm;
    }
    *(h8v*)(s_a1 + p * 32 + cb) = m.v;
  }
  __syncthreads();

  // ---- phase 2: conv2 as WMMA implicit GEMM ----
  // K permuted to (ky,kx,ic): B fragment = ONE contiguous 32B LDS load with
  // compile-time (ky*12+kx)*32 offsets. M=64 (4 tiles), N=64 (16/wave).
  const int n2 = wv * 16 + col;                         // spatial = oy*8+ox
  const _Float16* bbase = s_a1 + ((n2 >> 3) * 12 + (n2 & 7)) * 32 + hi * 16;
  union { f8v v; float f[8]; } acc[4];
#pragma unroll
  for (int mt = 0; mt < 4; ++mt)
#pragma unroll
    for (int r = 0; r < 8; ++r) acc[mt].f[r] = 0.f;

  for (int ky = 0; ky < 5; ++ky) {
#pragma unroll
    for (int kx = 0; kx < 5; ++kx) {
      h16v B = *(const h16v*)(bbase + (ky * 12 + kx) * 32);
      const int t = ky * 5 + kx;
#pragma unroll
      for (int mt = 0; mt < 4; ++mt) {
        union { h16v v; h8v p[2]; } A;
        const _Float16* wr = w2h + (mt * 16 + col) * 800 + t * 32 + hi * 16;
        A.p[0] = *(const h8v*)wr;
        A.p[1] = *(const h8v*)(wr + 8);
        acc[mt].v = __builtin_amdgcn_wmma_f32_16x16x32_f16(false, A.v, false, B,
                                                           (short)0, acc[mt].v, false, false);
      }
    }
  }
  // bn2 + relu -> s_a2 [spatial][ch] (aliases s_c1, dead by now)
#pragma unroll
  for (int mt = 0; mt < 4; ++mt) {
    union { f8v v; float f[8]; } o;
#pragma unroll
    for (int r = 0; r < 8; ++r) {
      int ch = mt * 16 + hi * 8 + r;
      float v = acc[mt].f[r] * sc[64 + ch] + sc[128 + ch];
      o.f[r] = v > 0.f ? v : 0.f;
    }
    *(f8v*)(s_a2 + n2 * 64 + mt * 16 + hi * 8) = o.v;   // one 32B DS store
  }
  __syncthreads();

  // ---- phase 3: 2x2 maxpool -> act2[img][c*16 + py*4 + px] (NCHW flatten)
  for (int o = tid; o < 1024; o += 128) {
    int c = o >> 4, p = o & 15, py = p >> 2, px = p & 3;
    float m00 = s_a2[((2 * py) * 8 + 2 * px) * 64 + c];
    float m01 = s_a2[((2 * py) * 8 + 2 * px + 1) * 64 + c];
    float m10 = s_a2[((2 * py + 1) * 8 + 2 * px) * 64 + c];
    float m11 = s_a2[((2 * py + 1) * 8 + 2 * px + 1) * 64 + c];
    act2[img * 1024 + o] = (_Float16)fmaxf(fmaxf(m00, m01), fmaxf(m10, m11));
  }
}

// ---------------- FC3: [8192,1024] x [512,1024]^T, bn+relu, f16 out ----
// each wave: 64(M) x 16(N) tile, K=1024 in 32 WMMA steps
__global__ __launch_bounds__(256) void k_fc3(
    const _Float16* __restrict__ act2, const _Float16* __restrict__ w3h,
    const float* __restrict__ sc, _Float16* __restrict__ act3)
{
  const int wv   = __builtin_amdgcn_readfirstlane(threadIdx.x >> 5); // scalar
  const int lane = threadIdx.x & 31;
  const int col  = lane & 15;
  const int hi   = lane >> 4;
  const int mbase = wv * 64;              // scalar M base: SGPR-based A stream
  const int n     = blockIdx.x * 16 + col; // all 8 waves share the same 16 rows

  union { f8v v; float f[8]; } acc[4];
#pragma unroll
  for (int mt = 0; mt < 4; ++mt)
#pragma unroll
    for (int r = 0; r < 8; ++r) acc[mt].f[r] = 0.f;

  const _Float16* brow = act2 + n * 1024;
#pragma unroll 2
  for (int k0 = 0; k0 < 1024; k0 += 32) {
    // keep the streaming B read one cache-tile ahead (global_prefetch_b8)
    __builtin_prefetch(brow + ((k0 + 256) & 1023), 0, 1);
    h16v B = *(const h16v*)(brow + k0 + hi * 16);   // 32B aligned global load
#pragma unroll
    for (int mt = 0; mt < 4; ++mt) {
      union { h16v v; h8v p[2]; } A;
      const _Float16* wr = w3h + (mbase + mt * 16 + col) * 1024 + k0 + hi * 16;
      A.p[0] = *(const h8v*)wr;
      A.p[1] = *(const h8v*)(wr + 8);
      acc[mt].v = __builtin_amdgcn_wmma_f32_16x16x32_f16(false, A.v, false, B,
                                                         (short)0, acc[mt].v, false, false);
    }
  }
#pragma unroll
  for (int mt = 0; mt < 4; ++mt) {
    int ch0 = mbase + mt * 16 + hi * 8;
    union { h8v v; _Float16 h[8]; } o;
#pragma unroll
    for (int r = 0; r < 8; ++r) {
      float v = acc[mt].f[r] * sc[192 + ch0 + r] + sc[704 + ch0 + r];
      o.h[r] = (_Float16)(v > 0.f ? v : 0.f);
    }
    *(h8v*)(act3 + n * 512 + ch0) = o.v;   // 8 contiguous channels per lane
  }
}

// ---------------- FC4: [8192,512] x [10,512]^T -> f32 out --------------
__global__ __launch_bounds__(256) void k_fc4(
    const _Float16* __restrict__ act3, const float* __restrict__ w4,
    float* __restrict__ out)
{
  int i = blockIdx.x * 256 + threadIdx.x;
  if (i >= 8192 * 10) return;
  int b = i / 10, c = i - b * 10;
  const _Float16* a = act3 + b * 512;
  const float*    w = w4 + c * 512;
  float s = 0.f;
  for (int k0 = 0; k0 < 512; k0 += 8) {
    h8v av = *(const h8v*)(a + k0);
#pragma unroll
    for (int j = 0; j < 8; ++j) s += (float)av[j] * w[k0 + j];
  }
  out[i] = s;
}

// ---------------- launch ----------------------------------------------
extern "C" void kernel_launch(void* const* d_in, const int* in_sizes, int n_in,
                              void* d_out, int out_size, void* d_ws, size_t ws_size,
                              hipStream_t stream)
{
  (void)in_sizes; (void)n_in; (void)out_size; (void)ws_size;
  const float* x   = (const float*)d_in[0];
  const float* w1  = (const float*)d_in[1];
  const float* b1  = (const float*)d_in[2];
  const float* g1  = (const float*)d_in[3];
  const float* be1 = (const float*)d_in[4];
  const float* m1  = (const float*)d_in[5];
  const float* v1  = (const float*)d_in[6];
  const float* w2  = (const float*)d_in[7];
  const float* b2  = (const float*)d_in[8];
  const float* g2  = (const float*)d_in[9];
  const float* be2 = (const float*)d_in[10];
  const float* m2  = (const float*)d_in[11];
  const float* v2  = (const float*)d_in[12];
  const float* w3  = (const float*)d_in[13];
  const float* b3  = (const float*)d_in[14];
  const float* g3  = (const float*)d_in[15];
  const float* be3 = (const float*)d_in[16];
  const float* m3  = (const float*)d_in[17];
  const float* v3  = (const float*)d_in[18];
  const float* w4  = (const float*)d_in[19];

  char* ws = (char*)d_ws;
  _Float16* w2h  = (_Float16*)(ws + WS_W2H);
  _Float16* w3h  = (_Float16*)(ws + WS_W3H);
  float*    scb  = (float*)(ws + WS_SC);
  _Float16* act2 = (_Float16*)(ws + WS_ACT2);
  _Float16* act3 = (_Float16*)(ws + WS_ACT3);

  k_prep_w <<<2048, 256, 0, stream>>>(w2, w3, w2h, w3h);           // 512K threads
  k_prep_bn<<<3,    256, 0, stream>>>(b1, g1, be1, m1, v1,
                                      b2, g2, be2, m2, v2,
                                      b3, g3, be3, m3, v3, scb);
  k_conv12 <<<8192, 128, 0, stream>>>(x, w1, w2h, scb, act2);      // 1 WG / image
  k_fc3    <<<512,  256, 0, stream>>>(act2, w3h, scb, act3);       // 4096 wave-tiles
  k_fc4    <<<320,  256, 0, stream>>>(act3, w4, (float*)d_out);
}
